// KVCacheAttention_78305843741188
// MI455X (gfx1250) — matrix operand order repaired and checked by reading, not compiled, not run on hardware
//
#include <hip/hip_runtime.h>
#include <hip/hip_bf16.h>

typedef __attribute__((ext_vector_type(16))) __bf16 v16bf;
typedef __attribute__((ext_vector_type(8)))  float  v8f;
typedef __attribute__((ext_vector_type(4)))  unsigned int u32x4;
typedef __attribute__((ext_vector_type(8)))  int i32x8;
typedef __attribute__((ext_vector_type(4)))  int i32x4;
typedef int gccv4i __attribute__((__vector_size__(16)));  // matches builtin param

#define D_MODEL 2048
#define NUM_HEADS 16
#define D_K 128
#define BATCH 2
#define S_NEW 1024
#define S_CACHE 3072
#define S_TOT 4096
#define M_ROWS (BATCH * S_NEW)   // 2048

// ---------------- CDNA5 async global->LDS path (ASYNCcnt) --------------------
#if __has_builtin(__builtin_amdgcn_global_load_async_to_lds_b128)
#define HAVE_ASYNC_LDS 1
__device__ __forceinline__ void async_cp16(const void* g, void* l) {
    __builtin_amdgcn_global_load_async_to_lds_b128(
        (__attribute__((address_space(1))) gccv4i*)g,
        (__attribute__((address_space(3))) gccv4i*)l, 0, 0);
}
__device__ __forceinline__ void async_wait0() {
#if __has_builtin(__builtin_amdgcn_s_wait_asynccnt)
    __builtin_amdgcn_s_wait_asynccnt(0);
#else
    asm volatile("s_wait_asynccnt 0x0" ::: "memory");
#endif
}
#else
#define HAVE_ASYNC_LDS 0
__device__ __forceinline__ void async_cp16(const void* g, void* l) {
    *(uint4*)l = *(const uint4*)g;
}
__device__ __forceinline__ void async_wait0() {}
#endif

// ---------------- CDNA5 Tensor Data Mover path (TENSORcnt) -------------------
#if __has_builtin(__builtin_amdgcn_tensor_load_to_lds) && \
    __has_builtin(__builtin_amdgcn_s_wait_tensorcnt)
#define HAVE_TDM 1
#if __has_include(<hip/amd_detail/amd_gfx1250_TDM.h>)
#define TDM_6ARG 1  // therock headers -> clang-23 6-arg builtin
#endif
#endif

// ---------------- fragment loaders (LDS, layouts per CDNA5 ISA 7.12.2) -------
// A-matrix 16x32 bf16: lane l holds row (l&15); K chunks at (l>>4)*8 and +16.
__device__ __forceinline__ v16bf frag_a(const __bf16* row_ptr, int koff) {
    union { v16bf v; uint4 u[2]; } f;
    f.u[0] = *(const uint4*)(row_ptr + koff);
    f.u[1] = *(const uint4*)(row_ptr + koff + 16);
    return f.v;
}
// B-matrix 32x16 bf16: lane l holds column (l&15); K chunks at (l>>4)*8 and +16.
__device__ __forceinline__ v16bf frag_b(const __bf16* row_ptr, int koff) {
    union { v16bf v; uint4 u[2]; } f;
    f.u[0] = *(const uint4*)(row_ptr + koff);
    f.u[1] = *(const uint4*)(row_ptr + koff + 16);
    return f.v;
}

__device__ __forceinline__ v8f wmma_bf16(v16bf a, v16bf b, v8f c) {
    return __builtin_amdgcn_wmma_f32_16x16x32_bf16(false, a, false, b, (short)0, c,
                                                   false, false);
}

// ---------------- fp32 -> bf16 convert ---------------------------------------
__global__ void cvt_bf16_kernel(const float* __restrict__ in,
                                __bf16* __restrict__ out, int n) {
    for (int i = blockIdx.x * blockDim.x + threadIdx.x; i < n;
         i += gridDim.x * blockDim.x)
        out[i] = (__bf16)in[i];
}

// ---------------- cache expand: fp32 cache -> out cache region + bf16 KV -----
__global__ void cache_expand_kernel(const float* __restrict__ cache,
                                    float* __restrict__ out_cache,
                                    __bf16* __restrict__ kb,
                                    __bf16* __restrict__ vb) {
    const int n = 2 * BATCH * NUM_HEADS * S_CACHE * D_K;
    for (int i = blockIdx.x * blockDim.x + threadIdx.x; i < n;
         i += gridDim.x * blockDim.x) {
        int d  = i & (D_K - 1);
        int t1 = i >> 7;
        int s  = t1 % S_CACHE;
        int t2 = t1 / S_CACHE;
        int h  = t2 & (NUM_HEADS - 1);
        int t3 = t2 >> 4;
        int b  = t3 & 1;
        int kv = t3 >> 1;
        float val = cache[i];
        size_t dst = ((size_t)((kv * BATCH + b) * NUM_HEADS + h) * S_TOT + s) * D_K + d;
        out_cache[dst] = val;
        __bf16* bbuf = kv ? vb : kb;
        bbuf[((size_t)(b * NUM_HEADS + h) * S_TOT + s) * D_K + d] = (__bf16)val;
    }
}

// ---------------- tiled bf16 WMMA GEMM:  C[m,n] = A[m,:] . W[n,:] + bias[n] ---
// Double-buffered through LDS with async global->LDS copies.
// MODE 0: Q  -> bf16 split-heads [b,h,s,d]
// MODE 1: K  -> fp32 cache region (s+3072) + bf16 K buffer
// MODE 2: V  -> fp32 cache region (s+3072) + bf16 V buffer
// MODE 3: O  -> fp32 plain [m,n]
#define LDG 40   // padded LDS stride (bf16) for 128x32 tiles
template <int MODE>
__global__ __launch_bounds__(256) void gemm_bf16_kernel(
    const __bf16* __restrict__ A, const __bf16* __restrict__ W,
    const float* __restrict__ bias, float* __restrict__ out_f32,
    __bf16* __restrict__ out_b16) {
    __shared__ __bf16 As[2][128 * LDG];
    __shared__ __bf16 Bs[2][128 * LDG];

    const int t    = threadIdx.x;
    const int lane = t & 31;
    const int w    = t >> 5;
    const int r    = lane & 15;
    const int half = lane >> 4;
    const int bm   = blockIdx.y * 128;
    const int bn   = blockIdx.x * 128;
    const int wm   = (w >> 2) * 64;  // 2 waves along M
    const int wn   = (w & 3) * 32;   // 4 waves along N

    auto load_tiles = [&](int k0, int buf) {
#pragma unroll
        for (int i = 0; i < 2; ++i) {
            int c = t + i * 256;          // 512 16B-chunks per tile
            int row = c >> 2, kc = c & 3;
            async_cp16(&A[(size_t)(bm + row) * D_MODEL + k0 + kc * 8],
                       &As[buf][row * LDG + kc * 8]);
            async_cp16(&W[(size_t)(bn + row) * D_MODEL + k0 + kc * 8],
                       &Bs[buf][row * LDG + kc * 8]);
        }
    };

    v8f acc[4][2];
#pragma unroll
    for (int mi = 0; mi < 4; ++mi)
#pragma unroll
        for (int ni = 0; ni < 2; ++ni) acc[mi][ni] = (v8f)(0.0f);

    load_tiles(0, 0);
    async_wait0();
    __syncthreads();

    int cur = 0;
    for (int k0 = 0; k0 < D_MODEL; k0 += 32) {
        if (k0 + 32 < D_MODEL) load_tiles(k0 + 32, cur ^ 1);  // prefetch next tile

        v16bf af[4], bf[2];
#pragma unroll
        for (int mi = 0; mi < 4; ++mi)
            af[mi] = frag_a(&As[cur][(wm + mi * 16 + r) * LDG], half * 8);
#pragma unroll
        for (int ni = 0; ni < 2; ++ni)
            bf[ni] = frag_b(&Bs[cur][(wn + ni * 16 + r) * LDG], half * 8);
#pragma unroll
        for (int mi = 0; mi < 4; ++mi)
#pragma unroll
            for (int ni = 0; ni < 2; ++ni)
                acc[mi][ni] = wmma_bf16(af[mi], bf[ni], acc[mi][ni]);

        async_wait0();
        __syncthreads();
        cur ^= 1;
    }

    // Epilogue. C layout: VGPR v, lanes 0-15: M=v,N=lane; lanes 16-31: M=v+8.
#pragma unroll
    for (int mi = 0; mi < 4; ++mi)
#pragma unroll
        for (int ni = 0; ni < 2; ++ni)
#pragma unroll
            for (int v = 0; v < 8; ++v) {
                int row = bm + wm + mi * 16 + v + 8 * half;  // (b,s)
                int col = bn + wn + ni * 16 + r;             // (h,d)
                float val = acc[mi][ni][v] + bias[col];
                int b = row >> 10, s = row & 1023;
                int h = col >> 7,  d = col & 127;
                if (MODE == 0) {
                    out_b16[((size_t)(b * NUM_HEADS + h) * S_NEW + s) * D_K + d] =
                        (__bf16)val;
                } else if (MODE == 1 || MODE == 2) {
                    int kv = MODE - 1;
                    out_f32[((size_t)((kv * BATCH + b) * NUM_HEADS + h) * S_TOT +
                             (S_CACHE + s)) * D_K + d] = val;
                    out_b16[((size_t)(b * NUM_HEADS + h) * S_TOT + (S_CACHE + s)) *
                                D_K + d] = (__bf16)val;
                } else {
                    out_f32[(size_t)row * D_MODEL + col] = val;
                }
            }
}

// ---------------- flash attention: 128 q-rows / WG, 128-key tiles ------------
#define LDA 136  // padded LDS stride for 128-wide bf16 rows (multiple of 8)
__global__ __launch_bounds__(256) void attn_kernel(
    const __bf16* __restrict__ qb, const __bf16* __restrict__ kb,
    const __bf16* __restrict__ vb, __bf16* __restrict__ attb) {
    extern __shared__ __align__(16) char smem[];
    __bf16* Qs = (__bf16*)smem;          // 128 x LDA
    __bf16* Ks = Qs + 128 * LDA;         // 128 x LDA
    __bf16* Vt = Ks + 128 * LDA;         // 128(d) x LDA(keys), transposed
    __bf16* Ps = Vt + 128 * LDA;         // 8 waves x 16 x LDA

    const int t    = threadIdx.x;
    const int lane = t & 31;
    const int w    = t >> 5;
    const int r    = lane & 15;
    const int half = lane >> 4;
    const int qt   = blockIdx.x;         // 0..7
    const int bh   = blockIdx.y;         // 0..31
    const int qbase = qt * 128;
    const float scale = 0.08838834764831845f;  // 1/sqrt(128)

    const __bf16* Qg = qb + (size_t)bh * S_NEW * D_K;
    const __bf16* Kg = kb + (size_t)bh * S_TOT * D_K;
    const __bf16* Vg = vb + (size_t)bh * S_TOT * D_K;

#pragma unroll
    for (int i = 0; i < 8; ++i) {
        int c = t + i * 256;
        int row = c >> 4, dc = c & 15;
        async_cp16(&Qg[(size_t)(qbase + row) * D_K + dc * 8],
                   &Qs[row * LDA + dc * 8]);
    }
    async_wait0();
    __syncthreads();

    v16bf qf[4];
#pragma unroll
    for (int kc = 0; kc < 4; ++kc)
        qf[kc] = frag_a(&Qs[(w * 16 + r) * LDA], half * 8 + kc * 32);

    v8f S[8], O[8];
    float m[8], l[8];
#pragma unroll
    for (int v = 0; v < 8; ++v) { m[v] = -1e30f; l[v] = 0.0f; O[v] = (v8f)(0.0f); }

    const int nkt = (qt + 25 < 32) ? (qt + 25) : 32;  // skip fully-masked tiles
    for (int kt = 0; kt < nkt; ++kt) {
        const int kbase = kt * 128;

#if defined(HAVE_TDM)
        // --- Tensor Data Mover: one descriptor moves the 128x128 bf16 K tile,
        //     pad_interval=64 DWORDs (=128 bf16), pad_amount=4 DWORDs (=8 bf16)
        //     reproduces the padded LDS stride of 136 elements.
        if (w == 0) {
            unsigned long long ga =
                (unsigned long long)(const void*)(Kg + (size_t)kbase * D_K);
            unsigned lds_off = (unsigned)(size_t)(
                (__attribute__((address_space(3))) char*)(void*)Ks);
            u32x4 g0;
            g0.x = 1u;                                     // count=1
            g0.y = lds_off;                                // lds_addr
            g0.z = (unsigned)ga;                           // global_addr[31:0]
            g0.w = (unsigned)((ga >> 32) & 0x01FFFFFFu) | (2u << 30);  // type=2
            i32x8 g1;
            g1[0] = (int)((1u << 16) |      // data_size = 2 bytes
                          (1u << 20) |      // pad_enable
                          (5u << 22) |      // pad_interval: 64 DWORDs
                          (3u << 25));      // pad_amount: 4 DWORDs
            g1[1] = (int)(128u << 16);      // tensor_dim0 = 128 (bits 79:48)
            g1[2] = (int)(4096u << 16);     // tensor_dim1 = 4096 (bits 111:80)
            g1[3] = (int)(128u << 16);      // tile_dim0 = 128 (bits 127:112)
            g1[4] = 128;                    // tile_dim1 = 128, tile_dim2 = 0
            g1[5] = 128;                    // tensor_dim0_stride = 128
            g1[6] = 0;
            g1[7] = 0;
            i32x4 z4 = {0, 0, 0, 0};
#if defined(TDM_6ARG)
            i32x8 z8 = {0, 0, 0, 0, 0, 0, 0, 0};
            __builtin_amdgcn_tensor_load_to_lds(g0, g1, z4, z4, z8, 0);
#else
            __builtin_amdgcn_tensor_load_to_lds(g0, g1, z4, z4, 0);
#endif
        }
#endif

#pragma unroll
        for (int i = 0; i < 8; ++i) {
            int c = t + i * 256;
            int row = c >> 4, dc = c & 15;
#if !defined(HAVE_TDM)
            async_cp16(&Kg[(size_t)(kbase + row) * D_K + dc * 8],
                       &Ks[row * LDA + dc * 8]);
#endif
            union { uint4 u; __bf16 e[8]; } vv;   // V needs lane-level transpose
            vv.u = *(const uint4*)&Vg[(size_t)(kbase + row) * D_K + dc * 8];
#pragma unroll
            for (int j = 0; j < 8; ++j) Vt[(dc * 8 + j) * LDA + row] = vv.e[j];
        }
#if defined(HAVE_TDM)
        if (w == 0) __builtin_amdgcn_s_wait_tensorcnt(0);
#else
        async_wait0();
#endif
        __syncthreads();

        // scores: S[nt] = Q . K^T for 8 key sub-tiles
#pragma unroll
        for (int nt = 0; nt < 8; ++nt) {
            v8f a = (v8f)(0.0f);
#pragma unroll
            for (int kc = 0; kc < 4; ++kc)
                a = wmma_bf16(qf[kc],
                              frag_b(&Ks[(nt * 16 + r) * LDA], half * 8 + kc * 32),
                              a);
            S[nt] = a;
        }

        // online softmax (row = v + 8*half per lane; cols across 16 lanes x 8 tiles)
#pragma unroll
        for (int v = 0; v < 8; ++v) {
            int qg = qbase + w * 16 + v + 8 * half;  // new-token index
            float mt = -1e30f;
#pragma unroll
            for (int nt = 0; nt < 8; ++nt) {
                float sv = S[nt][v] * scale;
                int j = kbase + nt * 16 + r;
                if (j >= qg + S_CACHE) sv = -1e30f;  // triu(k = s_total - s)
                S[nt][v] = sv;
                mt = fmaxf(mt, sv);
            }
#pragma unroll
            for (int off = 1; off < 16; off <<= 1)
                mt = fmaxf(mt, __shfl_xor(mt, off, 32));
            float mn = fmaxf(m[v], mt);
            float corr = __expf(m[v] - mn);
            float rs = 0.0f;
#pragma unroll
            for (int nt = 0; nt < 8; ++nt) {
                float p = __expf(S[nt][v] - mn);
                S[nt][v] = p;
                rs += p;
            }
#pragma unroll
            for (int off = 1; off < 16; off <<= 1) rs += __shfl_xor(rs, off, 32);
            l[v] = l[v] * corr + rs;
            m[v] = mn;
#pragma unroll
            for (int nt = 0; nt < 8; ++nt) O[nt][v] *= corr;
        }

        // stage P through per-wave LDS into A-fragment layout
        __bf16* Pw = Ps + w * 16 * LDA;
#pragma unroll
        for (int nt = 0; nt < 8; ++nt)
#pragma unroll
            for (int v = 0; v < 8; ++v)
                Pw[(v + 8 * half) * LDA + nt * 16 + r] = (__bf16)S[nt][v];
        __syncthreads();

        // O += P . V
#pragma unroll
        for (int dt = 0; dt < 8; ++dt)
#pragma unroll
            for (int kc = 0; kc < 4; ++kc) {
                v16bf pa = frag_a(&Pw[r * LDA], half * 8 + kc * 32);
                v16bf vf = frag_b(&Vt[(dt * 16 + r) * LDA], half * 8 + kc * 32);
                O[dt] = wmma_bf16(pa, vf, O[dt]);
            }
        __syncthreads();
    }

    // normalize + merge heads -> att[b, s, h*128+d] (bf16 for O-projection GEMM)
    const int b = bh >> 4, h = bh & 15;
#pragma unroll
    for (int v = 0; v < 8; ++v) {
        float inv = 1.0f / l[v];
        int srow = qbase + w * 16 + v + 8 * half;
#pragma unroll
        for (int dt = 0; dt < 8; ++dt)
            attb[((size_t)(b * S_NEW + srow)) * D_MODEL + h * D_K + dt * 16 + r] =
                (__bf16)(O[dt][v] * inv);
    }
}

// ---------------- host launcher ----------------------------------------------
extern "C" void kernel_launch(void* const* d_in, const int* in_sizes, int n_in,
                              void* d_out, int out_size, void* d_ws, size_t ws_size,
                              hipStream_t stream) {
    const float* x     = (const float*)d_in[0];
    const float* cache = (const float*)d_in[1];
    const float* Wq    = (const float*)d_in[2];
    const float* bq    = (const float*)d_in[3];
    const float* Wk    = (const float*)d_in[4];
    const float* bk    = (const float*)d_in[5];
    const float* Wv    = (const float*)d_in[6];
    const float* bv    = (const float*)d_in[7];
    const float* Wo    = (const float*)d_in[8];
    const float* bo    = (const float*)d_in[9];

    float* out       = (float*)d_out;
    float* out_cache = out + (size_t)BATCH * S_NEW * D_MODEL;  // cache_updated region

    char* ws = (char*)d_ws;
    size_t off = 0;
    auto alloc_bf16 = [&](size_t elems) {
        __bf16* p = (__bf16*)(ws + off);
        off += ((elems * 2 + 255) / 256) * 256;
        return p;
    };
    const size_t NW = (size_t)D_MODEL * D_MODEL;       // 4,194,304
    const size_t NX = (size_t)M_ROWS * D_MODEL;        // 4,194,304
    const size_t NKV = (size_t)BATCH * NUM_HEADS * S_TOT * D_K;  // 16,777,216

    __bf16* xb   = alloc_bf16(NX);
    __bf16* wqb  = alloc_bf16(NW);
    __bf16* wkb  = alloc_bf16(NW);
    __bf16* wvb  = alloc_bf16(NW);
    __bf16* wob  = alloc_bf16(NW);
    __bf16* qbuf = alloc_bf16(NX);
    __bf16* kbuf = alloc_bf16(NKV);
    __bf16* vbuf = alloc_bf16(NKV);
    __bf16* attb = alloc_bf16(NX);

    cvt_bf16_kernel<<<2048, 256, 0, stream>>>(x, xb, (int)NX);
    cvt_bf16_kernel<<<2048, 256, 0, stream>>>(Wq, wqb, (int)NW);
    cvt_bf16_kernel<<<2048, 256, 0, stream>>>(Wk, wkb, (int)NW);
    cvt_bf16_kernel<<<2048, 256, 0, stream>>>(Wv, wvb, (int)NW);
    cvt_bf16_kernel<<<2048, 256, 0, stream>>>(Wo, wob, (int)NW);

    cache_expand_kernel<<<4096, 256, 0, stream>>>(cache, out_cache, kbuf, vbuf);

    dim3 gg(D_MODEL / 128, M_ROWS / 128);  // 16 x 16
    gemm_bf16_kernel<0><<<gg, 256, 0, stream>>>(xb, wqb, bq, nullptr, qbuf);
    gemm_bf16_kernel<1><<<gg, 256, 0, stream>>>(xb, wkb, bk, out_cache, kbuf);
    gemm_bf16_kernel<2><<<gg, 256, 0, stream>>>(xb, wvb, bv, out_cache, vbuf);

    size_t shmem = (size_t)(3 * 128 * LDA + 8 * 16 * LDA) * sizeof(__bf16);  // 139264
    attn_kernel<<<dim3(8, BATCH * NUM_HEADS), 256, shmem, stream>>>(qbuf, kbuf,
                                                                    vbuf, attb);

    gemm_bf16_kernel<3><<<gg, 256, 0, stream>>>(attb, wob, bo, out, nullptr);
}
